// LinearRNNBlock_71262097375477
// MI455X (gfx1250) — compile-verified, matrix operand order checked by reading, not run on hardware
//
#include <hip/hip_runtime.h>
#include <hip/hip_bf16.h>
#include <math.h>

typedef __attribute__((ext_vector_type(16))) __bf16 v16bf;
typedef __attribute__((ext_vector_type(8)))  __bf16 v8bf;
typedef __attribute__((ext_vector_type(8)))  float  v8f;

#define BSZ 8
#define TSZ 2048
#define CSZ 1024
#define NTOK (BSZ*TSZ)
#define NCHUNK 64
#define TCH (TSZ/NCHUNK)
#define EPSV 1e-6f

// low 32 bits of a generic pointer to __shared__ = LDS byte offset (flat LDS aperture)
__device__ __forceinline__ unsigned lds_off(const void* p) {
  return (unsigned)(unsigned long long)p;
}

// gfx1250 async copy: global memory -> LDS, 16B per lane, tracked by ASYNCcnt.
// GVS mode: 64-bit uniform base (SGPR pair) + 32-bit per-lane byte offset.
__device__ __forceinline__ void async_cp_b128(unsigned ldsaddr, unsigned goff, const void* base) {
  asm volatile("global_load_async_to_lds_b128 %0, %1, %2"
               :
               : "v"(ldsaddr), "v"(goff), "s"(base)
               : "memory");
}
__device__ __forceinline__ void wait_async0() {
  asm volatile("s_wait_asynccnt 0x0" ::: "memory");
}

// ---------------- rstd per token: rsqrt(mean(x^2)+eps) ----------------
__global__ __launch_bounds__(256) void rstd_kernel(const float* __restrict__ src,
                                                   float* __restrict__ rstd) {
  int token = blockIdx.x;
  const float* p = src + (size_t)token * CSZ;
  float s = 0.f;
  for (int i = threadIdx.x; i < CSZ; i += 256) { float v = p[i]; s += v * v; }
  #pragma unroll
  for (int off = 16; off > 0; off >>= 1) s += __shfl_xor(s, off, 32);
  __shared__ float red[8];
  if ((threadIdx.x & 31) == 0) red[threadIdx.x >> 5] = s;
  __syncthreads();
  if (threadIdx.x == 0) {
    float t = 0.f;
    #pragma unroll
    for (int i = 0; i < 8; ++i) t += red[i];
    rstd[token] = rsqrtf(t * (1.0f / CSZ) + EPSV);
  }
}

// ---------------- per-chunk partial sums of y = x*rstd*w ----------------
__global__ __launch_bounds__(256) void chunk_sum_kernel(const float* __restrict__ x,
                                                        const float* __restrict__ rstd,
                                                        const float* __restrict__ w,
                                                        float* __restrict__ csum) {
  int idx = blockIdx.x * 256 + threadIdx.x;          // [b][chunk][c]
  int c = idx % CSZ;
  int chunk = (idx / CSZ) % NCHUNK;
  int b = idx / (CSZ * NCHUNK);
  float wc = w[c];
  size_t base = ((size_t)b * TSZ + (size_t)chunk * TCH) * CSZ + c;
  int rb = b * TSZ + chunk * TCH;
  float s = 0.f;
  #pragma unroll 4
  for (int j = 0; j < TCH; ++j) s += x[base + (size_t)j * CSZ] * rstd[rb + j] * wc;
  csum[idx] = s;
}

// ---------------- chunk prefix + in-chunk scan + triangular scaler -> bf16 state ----------------
__global__ __launch_bounds__(256) void state_kernel(const float* __restrict__ x,
                                                    const float* __restrict__ rstd,
                                                    const float* __restrict__ w,
                                                    const float* __restrict__ csum,
                                                    __bf16* __restrict__ state) {
  int idx = blockIdx.x * 256 + threadIdx.x;
  int c = idx % CSZ;
  int chunk = (idx / CSZ) % NCHUNK;
  int b = idx / (CSZ * NCHUNK);
  float wc = w[c];
  float run = 0.f;
  const float* cs = csum + (size_t)b * NCHUNK * CSZ + c;
  for (int k = 0; k < chunk; ++k) run += cs[(size_t)k * CSZ];
  size_t base = ((size_t)b * TSZ + (size_t)chunk * TCH) * CSZ + c;
  int rb = b * TSZ + chunk * TCH;
  for (int j = 0; j < TCH; ++j) {
    int t = chunk * TCH + j;
    run += x[base + (size_t)j * CSZ] * rstd[rb + j] * wc;
    float scl = 0.5f * (float)(t + 1) * (float)(t + 2);   // cumsum(arange(1..T))[t]
    state[base + (size_t)j * CSZ] = (__bf16)(run / scl);
  }
}

// ---------------- rmsnorm scale to bf16: dst = bf16(src * rstd[token] * w[c]) ----------------
__global__ __launch_bounds__(256) void scale_bf16_kernel(const float* __restrict__ src,
                                                         const float* __restrict__ rstd,
                                                         const float* __restrict__ w,
                                                         __bf16* __restrict__ dst) {
  int idx = blockIdx.x * 256 + threadIdx.x;
  int c = idx % CSZ;
  int token = idx / CSZ;
  dst[idx] = (__bf16)(src[idx] * rstd[token] * w[c]);
}

// ---------------- fp32 (K,N) -> bf16 (N,K) tiled transpose-convert ----------------
__global__ __launch_bounds__(256) void transpose_bf16_kernel(const float* __restrict__ src,
                                                             __bf16* __restrict__ dst,
                                                             int K, int N) {
  __shared__ float tile[32][33];
  int nB = blockIdx.x * 32, kB = blockIdx.y * 32;
  int tx = threadIdx.x, ty = threadIdx.y;  // 32 x 8
  #pragma unroll
  for (int i = 0; i < 32; i += 8) tile[ty + i][tx] = src[(size_t)(kB + ty + i) * N + nB + tx];
  __syncthreads();
  #pragma unroll
  for (int i = 0; i < 32; i += 8) dst[(size_t)(nB + ty + i) * K + kB + tx] = (__bf16)tile[tx][ty + i];
}

// ---------------- bf16 WMMA GEMM: C(MxN) = A(MxK) * Bt(NxK)^T, fused epilogues ----------------
// Global->LDS staging uses gfx1250 GLOBAL_LOAD_ASYNC_TO_LDS_B128 (ASYNCcnt-tracked DMA),
// double-buffered; compute is v_wmma_f32_16x16x32_bf16 from ds_load_b128 fragments.
// EPI 0: out = sigmoid(acc+bias) * aux      (fp32 out)
// EPI 1: out = bf16(relu(acc+bias))          (bf16 out)
// EPI 2: out = acc + bias + aux              (fp32 out, residual)
template <int EPI>
__global__ __launch_bounds__(256) void gemm_bf16_kernel(const __bf16* __restrict__ A,
                                                        const __bf16* __restrict__ Bt,
                                                        void* __restrict__ Cout,
                                                        const float* __restrict__ bias,
                                                        const float* __restrict__ aux,
                                                        int M, int N, int K) {
  __shared__ __attribute__((aligned(16))) __bf16 sA[2][128 * 40];  // 80B row stride
  __shared__ __attribute__((aligned(16))) __bf16 sB[2][128 * 40];

  const int tid  = threadIdx.x;
  const int lane = tid & 31;
  const int wave = tid >> 5;     // 8 waves
  const int wm   = wave & 1;     // 2 along M (64 rows each)
  const int wn   = wave >> 1;    // 4 along N (32 cols each)
  const int r    = lane & 15;
  const int halfk = lane >> 4;

  const int mBase = blockIdx.y * 128;
  const int nBase = blockIdx.x * 128;

  const int rowL = tid >> 2;     // 0..63 (loader row; +64 for second half)
  const int seg  = tid & 3;      // 16B segment within a 64B (32-elem) row

  // per-thread LDS destination byte addresses (buffer 0)
  const unsigned bufStride = 128u * 40u * 2u;  // 10240 B between double buffers
  const unsigned ldsA0 = lds_off(&sA[0][rowL * 40 + seg * 8]);
  const unsigned ldsA1 = lds_off(&sA[0][(rowL + 64) * 40 + seg * 8]);
  const unsigned ldsB0 = lds_off(&sB[0][rowL * 40 + seg * 8]);
  const unsigned ldsB1 = lds_off(&sB[0][(rowL + 64) * 40 + seg * 8]);

  // per-thread global byte offsets at k=0
  const unsigned rowSkip = (unsigned)((size_t)64 * K * 2);
  const unsigned ga = (unsigned)(((size_t)(mBase + rowL) * K + seg * 8) * 2);
  const unsigned gb = (unsigned)(((size_t)(nBase + rowL) * K + seg * 8) * 2);

  v8f acc[4][2] = {};
  const int KT = K >> 5;

  // stage tile 0 via async DMA
  async_cp_b128(ldsA0, ga, A);
  async_cp_b128(ldsA1, ga + rowSkip, A);
  async_cp_b128(ldsB0, gb, Bt);
  async_cp_b128(ldsB1, gb + rowSkip, Bt);
  wait_async0();
  __syncthreads();

  for (int kt = 0; kt < KT; ++kt) {
    const int cur = kt & 1;
    if (kt + 1 < KT) {  // DMA next tile into the other buffer; overlaps with WMMA below
      const unsigned koff = (unsigned)(kt + 1) << 6;  // 32 elems * 2B
      const unsigned lo = (unsigned)(cur ^ 1) * bufStride;
      async_cp_b128(ldsA0 + lo, ga + koff, A);
      async_cp_b128(ldsA1 + lo, ga + rowSkip + koff, A);
      async_cp_b128(ldsB0 + lo, gb + koff, Bt);
      async_cp_b128(ldsB1 + lo, gb + rowSkip + koff, Bt);
    }
    // fragment loads: lane L%16 = row, K-runs at 8*(L/16) and +16 (ISA 16-bit A layout)
    const __bf16* Ab = sA[cur];
    const __bf16* Bb = sB[cur];
    v16bf af[4], bfrag[2];
    #pragma unroll
    for (int n = 0; n < 2; ++n) {
      const __bf16* p = Bb + (wn * 32 + n * 16 + r) * 40 + halfk * 8;
      *((v8bf*)&bfrag[n])     = *(const v8bf*)(p);
      *((v8bf*)&bfrag[n] + 1) = *(const v8bf*)(p + 16);
    }
    #pragma unroll
    for (int m = 0; m < 4; ++m) {
      const __bf16* p = Ab + (wm * 64 + m * 16 + r) * 40 + halfk * 8;
      *((v8bf*)&af[m])     = *(const v8bf*)(p);
      *((v8bf*)&af[m] + 1) = *(const v8bf*)(p + 16);
    }
    #pragma unroll
    for (int m = 0; m < 4; ++m)
      #pragma unroll
      for (int n = 0; n < 2; ++n)
        acc[m][n] = __builtin_amdgcn_wmma_f32_16x16x32_bf16(
            false, af[m], false, bfrag[n], (short)0, acc[m][n], false, false);

    wait_async0();     // this wave's DMA into the other buffer is done
    __syncthreads();   // everyone's DMA done -> safe to read next buffer
  }

  // epilogue: C/D layout — lane holds col r (+16 cols via lane>>4 rows +8), VGPR e = row e
  #pragma unroll
  for (int m = 0; m < 4; ++m) {
    #pragma unroll
    for (int n = 0; n < 2; ++n) {
      int col  = nBase + wn * 32 + n * 16 + r;
      int row0 = mBase + wm * 64 + m * 16 + halfk * 8;
      float bv = bias[col];
      #pragma unroll
      for (int e = 0; e < 8; ++e) {
        size_t idx = (size_t)(row0 + e) * N + col;
        float v = acc[m][n][e] + bv;
        if (EPI == 0) {
          float g = 1.0f / (1.0f + __expf(-v));
          ((float*)Cout)[idx] = g * aux[idx];
        } else if (EPI == 1) {
          ((__bf16*)Cout)[idx] = (__bf16)fmaxf(v, 0.0f);
        } else {
          ((float*)Cout)[idx] = v + aux[idx];
        }
      }
    }
  }
}

extern "C" void kernel_launch(void* const* d_in, const int* in_sizes, int n_in,
                              void* d_out, int out_size, void* d_ws, size_t ws_size,
                              hipStream_t stream) {
  (void)in_sizes; (void)n_in; (void)out_size; (void)ws_size;
  const float* x   = (const float*)d_in[0];
  const float* n1w = (const float*)d_in[1];
  const float* w1  = (const float*)d_in[2];  // (C, C)
  const float* b1  = (const float*)d_in[3];
  const float* n2w = (const float*)d_in[4];
  const float* w2  = (const float*)d_in[5];  // (C, 4C)
  const float* b2  = (const float*)d_in[6];
  const float* w3  = (const float*)d_in[7];  // (4C, C)
  const float* b3  = (const float*)d_in[8];
  float* out = (float*)d_out;

  char* ws = (char*)d_ws;
  size_t off = 0;
  auto carve = [&](size_t bytes) -> char* {
    char* p = ws + off;
    off += (bytes + 255) & ~(size_t)255;
    return p;
  };

  float*  rstd1 = (float*)carve((size_t)NTOK * 4);
  float*  csum  = (float*)carve((size_t)BSZ * NCHUNK * CSZ * 4);
  __bf16* stB   = (__bf16*)carve((size_t)NTOK * CSZ * 2);
  __bf16* w1t   = (__bf16*)carve((size_t)CSZ * CSZ * 2);
  float*  out1  = (float*)carve((size_t)NTOK * CSZ * 4);
  float*  rstd2 = (float*)carve((size_t)NTOK * 4);
  __bf16* hB    = (__bf16*)carve((size_t)NTOK * CSZ * 2);
  __bf16* w2t   = (__bf16*)carve((size_t)4 * CSZ * CSZ * 2);
  __bf16* h1B   = (__bf16*)carve((size_t)NTOK * 4 * CSZ * 2);
  __bf16* w3t   = (__bf16*)carve((size_t)4 * CSZ * CSZ * 2);

  dim3 tb(32, 8);
  transpose_bf16_kernel<<<dim3(CSZ / 32, CSZ / 32), tb, 0, stream>>>(w1, w1t, CSZ, CSZ);
  transpose_bf16_kernel<<<dim3(4 * CSZ / 32, CSZ / 32), tb, 0, stream>>>(w2, w2t, CSZ, 4 * CSZ);
  transpose_bf16_kernel<<<dim3(CSZ / 32, 4 * CSZ / 32), tb, 0, stream>>>(w3, w3t, 4 * CSZ, CSZ);

  rstd_kernel<<<NTOK, 256, 0, stream>>>(x, rstd1);
  chunk_sum_kernel<<<(BSZ * NCHUNK * CSZ) / 256, 256, 0, stream>>>(x, rstd1, n1w, csum);
  state_kernel<<<(BSZ * NCHUNK * CSZ) / 256, 256, 0, stream>>>(x, rstd1, n1w, csum, stB);

  // gate = sigmoid(state @ w1 + b1); out1 = gate * x
  gemm_bf16_kernel<0><<<dim3(CSZ / 128, NTOK / 128), 256, 0, stream>>>(
      stB, w1t, out1, b1, x, NTOK, CSZ, CSZ);

  rstd_kernel<<<NTOK, 256, 0, stream>>>(out1, rstd2);
  scale_bf16_kernel<<<((size_t)NTOK * CSZ) / 256, 256, 0, stream>>>(out1, rstd2, n2w, hB);

  // h1 = relu(h @ w2 + b2)  (bf16)
  gemm_bf16_kernel<1><<<dim3(4 * CSZ / 128, NTOK / 128), 256, 0, stream>>>(
      hB, w2t, h1B, b2, nullptr, NTOK, 4 * CSZ, CSZ);

  // out = h1 @ w3 + b3 + out1
  gemm_bf16_kernel<2><<<dim3(CSZ / 128, NTOK / 128), 256, 0, stream>>>(
      h1B, w3t, out, b3, out1, NTOK, CSZ, 4 * CSZ);
}
